// SparseMoE_58136677318851
// MI455X (gfx1250) — matrix operand order repaired and checked by reading, not compiled, not run on hardware
//
#include <hip/hip_runtime.h>
#include <hip/hip_bf16.h>
#include <math.h>

// ---------------------------------------------------------------------------
// SparseMoE forward for gfx1250 (MI455X). bf16 WMMA compute, fp32 accumulate.
// Weights converted fp32->bf16 once per expert pass; GEMM hot loop is pure
// bf16 b128 loads + v_wmma_f32_16x16x32_bf16 (1:1 load:wmma ratio).
// ---------------------------------------------------------------------------

typedef __bf16 bf16_t;
typedef __attribute__((ext_vector_type(16))) bf16_t v16bf;
typedef __attribute__((ext_vector_type(8)))  bf16_t v8bf;
typedef __attribute__((ext_vector_type(4)))  bf16_t v4bf;
typedef __attribute__((ext_vector_type(8)))  float  v8f;
typedef __attribute__((ext_vector_type(4)))  float  f32x4;

#define TOK   4096
#define HDIM  1024
#define IDIM  4096
#define NEXP  8

union FragBF { v16bf v; v8bf h[2]; };

// ---- fragment loaders ------------------------------------------------------
// 16-bit A fragment, 16x32 (MxK), per ISA layout:
//   lane<16 : row=lane,    elems[0..7]=K[0..7],   elems[8..15]=K[16..23]
//   lane>=16: row=lane-16, elems[0..7]=K[8..15],  elems[8..15]=K[24..31]
__device__ inline v16bf load_a_frag(const bf16_t* __restrict__ base, int ld,
                                    int mbase, int k0, int lane) {
  const int r  = lane & 15;
  const int hi = lane >> 4;
  const bf16_t* p = base + (size_t)(mbase + r) * ld + k0 + hi * 8;
  FragBF u;
  u.h[0] = *(const v8bf*)(p);
  u.h[1] = *(const v8bf*)(p + 16);
  return u.v;
}

// B fragment, 32x16 (KxN): column n per lane; same K layout as A fragment.
// Weights are pre-converted bf16, row-major along K -> two b128 (K+8h, K+16+8h).
__device__ inline v16bf load_b_frag(const bf16_t* __restrict__ base, int ld,
                                    int nbase, int k0, int lane) {
  const int n  = lane & 15;
  const int hi = lane >> 4;
  const bf16_t* p = base + (size_t)(nbase + n) * ld + k0 + hi * 8;
  FragBF u;
  u.h[0] = *(const v8bf*)(p);
  u.h[1] = *(const v8bf*)(p + 16);
  return u.v;
}

__device__ inline v8f wmma_bf16(v16bf a, v16bf b, v8f c) {
  return __builtin_amdgcn_wmma_f32_16x16x32_bf16(
      /*neg_a=*/false, a, /*neg_b=*/false, b,
      /*c_mod=*/(short)0, c, /*reuse_a=*/false, /*reuse_b=*/false);
}

// ---- fp32 -> bf16 streaming conversion (x and per-expert weights) ----------
// 4 elements/thread; prefetch ahead on the streamed fp32 source.
__global__ __launch_bounds__(256) void cvt_to_bf16_v4(const float* __restrict__ src,
                                                      bf16_t* __restrict__ dst, int n4) {
  int i = blockIdx.x * blockDim.x + threadIdx.x;
  if (i < n4) {
    f32x4 f = ((const f32x4*)src)[i];
    v4bf o;
    o[0] = (bf16_t)f.x; o[1] = (bf16_t)f.y; o[2] = (bf16_t)f.z; o[3] = (bf16_t)f.w;
    ((v4bf*)dst)[i] = o;
    __builtin_prefetch((const void*)(src + (size_t)i * 4 + 16384), 0, 0);
  }
}

// ---- router: logits, top-2 softmax gates (coef), per-token aux pieces ------
__global__ __launch_bounds__(256) void moe_router(
    const float* __restrict__ x, const float* __restrict__ wg,
    float* __restrict__ logits_out, float* __restrict__ coef,
    float* __restrict__ probs_buf, int* __restrict__ idx_buf,
    float* __restrict__ z2_buf) {
  __shared__ float red[256][NEXP];
  const int t = blockIdx.x;
  const int tid = threadIdx.x;
  float part[NEXP];
#pragma unroll
  for (int e = 0; e < NEXP; ++e) part[e] = 0.f;
  const float* xr = x + (size_t)t * HDIM;
  for (int k = tid; k < HDIM; k += 256) {
    float xv = xr[k];
    const float* wr = wg + (size_t)k * NEXP;
#pragma unroll
    for (int e = 0; e < NEXP; ++e) part[e] += xv * wr[e];
  }
#pragma unroll
  for (int e = 0; e < NEXP; ++e) red[tid][e] = part[e];
  __syncthreads();
  for (int s = 128; s > 0; s >>= 1) {
    if (tid < s) {
#pragma unroll
      for (int e = 0; e < NEXP; ++e) red[tid][e] += red[tid + s][e];
    }
    __syncthreads();
  }
  if (tid == 0) {
    float lg[NEXP];
#pragma unroll
    for (int e = 0; e < NEXP; ++e) {
      lg[e] = red[0][e];
      logits_out[(size_t)t * NEXP + e] = lg[e];
    }
    float mx = lg[0];
#pragma unroll
    for (int e = 1; e < NEXP; ++e) mx = fmaxf(mx, lg[e]);
    float pe[NEXP], se = 0.f;
#pragma unroll
    for (int e = 0; e < NEXP; ++e) { pe[e] = expf(lg[e] - mx); se += pe[e]; }
#pragma unroll
    for (int e = 0; e < NEXP; ++e) probs_buf[(size_t)t * NEXP + e] = pe[e] / se;
    // top-2 (ties -> lowest index, matching jax.lax.top_k)
    int i0 = 0;
#pragma unroll
    for (int e = 1; e < NEXP; ++e) if (lg[e] > lg[i0]) i0 = e;
    int i1 = (i0 == 0) ? 1 : 0;
#pragma unroll
    for (int e = 0; e < NEXP; ++e) if (e != i0 && lg[e] > lg[i1]) i1 = e;
    float v0 = lg[i0], v1 = lg[i1];
    float m2 = fmaxf(v0, v1);
    float g0 = expf(v0 - m2), g1 = expf(v1 - m2);
    float gs = g0 + g1;
    g0 /= gs; g1 /= gs;
#pragma unroll
    for (int e = 0; e < NEXP; ++e) coef[(size_t)t * NEXP + e] = 0.f;
    coef[(size_t)t * NEXP + i0] = g0;
    coef[(size_t)t * NEXP + i1] = g1;
    idx_buf[2 * t]     = i0;
    idx_buf[2 * t + 1] = i1;
    float lse = mx + logf(se);
    z2_buf[t] = lse * lse;
  }
}

// ---- GEMM1: h = gelu(x @ w_fc[e]^T + b_fc[e]), bf16 out --------------------
// Block: 128 threads (4 waves), block tile 128x128, wave tile 64x64.
__global__ __launch_bounds__(128) void moe_fc_gelu(
    const bf16_t* __restrict__ xb, const bf16_t* __restrict__ wfcb,
    const float* __restrict__ bfc, bf16_t* __restrict__ hbuf) {
  const int lane = threadIdx.x & 31;
  const int wave = threadIdx.x >> 5;
  const int mbase = blockIdx.y * 128 + (wave >> 1) * 64;
  const int nbase = blockIdx.x * 128 + (wave & 1) * 64;
  v8f acc[4][4] = {};
  for (int k0 = 0; k0 < HDIM; k0 += 32) {
    v16bf a[4], b[4];
#pragma unroll
    for (int i = 0; i < 4; ++i) a[i] = load_a_frag(xb, HDIM, mbase + 16 * i, k0, lane);
#pragma unroll
    for (int j = 0; j < 4; ++j) b[j] = load_b_frag(wfcb, HDIM, nbase + 16 * j, k0, lane);
#pragma unroll
    for (int i = 0; i < 4; ++i)
#pragma unroll
      for (int j = 0; j < 4; ++j) acc[i][j] = wmma_bf16(a[i], b[j], acc[i][j]);
  }
  const int nlo = lane & 15;
  const int hi  = lane >> 4;
#pragma unroll
  for (int i = 0; i < 4; ++i)
#pragma unroll
    for (int j = 0; j < 4; ++j) {
      const int col = nbase + 16 * j + nlo;
      const float bias = bfc[col];
#pragma unroll
      for (int v = 0; v < 8; ++v) {
        const int row = mbase + 16 * i + v + 8 * hi;
        float xv = acc[i][j][v] + bias;
        float g = 0.5f * xv * (1.0f + erff(xv * 0.70710678118654752f));
        hbuf[(size_t)row * IDIM + col] = (bf16_t)g;
      }
    }
}

// ---- GEMM2: out (+)= coef[:,e] * (h @ w_proj[e]^T + b_proj[e]) -------------
__global__ __launch_bounds__(128) void moe_proj_combine(
    const bf16_t* __restrict__ hbuf, const bf16_t* __restrict__ wprojb,
    const float* __restrict__ bproj, const float* __restrict__ coef,
    int e, int first, float* __restrict__ out) {
  const int lane = threadIdx.x & 31;
  const int wave = threadIdx.x >> 5;
  const int mbase = blockIdx.y * 128 + (wave >> 1) * 64;
  const int nbase = blockIdx.x * 128 + (wave & 1) * 64;
  v8f acc[4][4] = {};
  for (int k0 = 0; k0 < IDIM; k0 += 32) {
    v16bf a[4], b[4];
#pragma unroll
    for (int i = 0; i < 4; ++i) a[i] = load_a_frag(hbuf, IDIM, mbase + 16 * i, k0, lane);
#pragma unroll
    for (int j = 0; j < 4; ++j) b[j] = load_b_frag(wprojb, IDIM, nbase + 16 * j, k0, lane);
#pragma unroll
    for (int i = 0; i < 4; ++i)
#pragma unroll
      for (int j = 0; j < 4; ++j) acc[i][j] = wmma_bf16(a[i], b[j], acc[i][j]);
  }
  const int nlo = lane & 15;
  const int hi  = lane >> 4;
#pragma unroll
  for (int i = 0; i < 4; ++i)
#pragma unroll
    for (int j = 0; j < 4; ++j) {
      const int col = nbase + 16 * j + nlo;
      const float bias = bproj[col];
#pragma unroll
      for (int v = 0; v < 8; ++v) {
        const int row = mbase + 16 * i + v + 8 * hi;
        const float c = coef[(size_t)row * NEXP + e];
        const float r = c * (acc[i][j][v] + bias);
        const size_t o = (size_t)row * HDIM + col;
        out[o] = first ? r : (out[o] + r);
      }
    }
}

// ---- deterministic aux-loss reduction --------------------------------------
__global__ __launch_bounds__(256) void moe_finalize(
    const float* __restrict__ probs_buf, const int* __restrict__ idx_buf,
    const float* __restrict__ z2_buf, float* __restrict__ aux_out) {
  __shared__ float red[256][17];
  const int tid = threadIdx.x;
  float a[17];
#pragma unroll
  for (int k = 0; k < 17; ++k) a[k] = 0.f;
  for (int t = tid; t < TOK; t += 256) {
#pragma unroll
    for (int e = 0; e < NEXP; ++e) a[e] += probs_buf[(size_t)t * NEXP + e];
    a[8 + idx_buf[2 * t]]     += 1.f;
    a[8 + idx_buf[2 * t + 1]] += 1.f;
    a[16] += z2_buf[t];
  }
#pragma unroll
  for (int k = 0; k < 17; ++k) red[tid][k] = a[k];
  __syncthreads();
  for (int s = 128; s > 0; s >>= 1) {
    if (tid < s) {
#pragma unroll
      for (int k = 0; k < 17; ++k) red[tid][k] += red[tid + s][k];
    }
    __syncthreads();
  }
  if (tid == 0) {
    float sp = 0.f, sf = 0.f;
#pragma unroll
    for (int e = 0; e < NEXP; ++e) { sp += fabsf(red[0][e]); sf += fabsf(red[0][8 + e]); }
    float sw = 0.f;
#pragma unroll
    for (int e = 0; e < NEXP; ++e) sw += (red[0][e] / sp) * (red[0][8 + e] / sf);
    sw *= (float)NEXP;
    aux_out[0] = sw + 0.1f * (red[0][16] / (float)TOK);
  }
}

// ---------------------------------------------------------------------------
extern "C" void kernel_launch(void* const* d_in, const int* in_sizes, int n_in,
                              void* d_out, int out_size, void* d_ws, size_t ws_size,
                              hipStream_t stream) {
  const float* x     = (const float*)d_in[0];  // [T, H]
  const float* wg    = (const float*)d_in[1];  // [H, E]
  const float* wfc   = (const float*)d_in[2];  // [E, I, H]
  const float* bfc   = (const float*)d_in[3];  // [E, I]
  const float* wproj = (const float*)d_in[4];  // [E, H, I]
  const float* bproj = (const float*)d_in[5];  // [E, H]

  float* out    = (float*)d_out;                     // [T*H]
  float* logits = out + (size_t)TOK * HDIM;          // [T*E]
  float* aux    = logits + (size_t)TOK * NEXP;       // [1]

  char* ws = (char*)d_ws;
  size_t off = 0;
  bf16_t* xb   = (bf16_t*)(ws + off); off += (size_t)TOK * HDIM * sizeof(bf16_t);  //  8 MB
  bf16_t* hbuf = (bf16_t*)(ws + off); off += (size_t)TOK * IDIM * sizeof(bf16_t);  // 32 MB
  bf16_t* wfcb = (bf16_t*)(ws + off); off += (size_t)IDIM * HDIM * sizeof(bf16_t); //  8 MB
  bf16_t* wprb = (bf16_t*)(ws + off); off += (size_t)HDIM * IDIM * sizeof(bf16_t); //  8 MB
  float* coef  = (float*)(ws + off);  off += (size_t)TOK * NEXP * sizeof(float);
  float* probs = (float*)(ws + off);  off += (size_t)TOK * NEXP * sizeof(float);
  float* z2    = (float*)(ws + off);  off += (size_t)TOK * sizeof(float);
  int*   idxb  = (int*)(ws + off);    off += (size_t)TOK * 2 * sizeof(int);

  const int nx4 = (TOK * HDIM) / 4;
  const int nw4 = (IDIM * HDIM) / 4;
  cvt_to_bf16_v4<<<(nx4 + 255) / 256, 256, 0, stream>>>(x, xb, nx4);
  moe_router<<<TOK, 256, 0, stream>>>(x, wg, logits, coef, probs, idxb, z2);

  dim3 g1(IDIM / 128, TOK / 128);  // 32 x 32
  dim3 g2(HDIM / 128, TOK / 128);  // 8 x 32
  for (int e = 0; e < NEXP; ++e) {
    cvt_to_bf16_v4<<<(nw4 + 255) / 256, 256, 0, stream>>>(
        wfc + (size_t)e * IDIM * HDIM, wfcb, nw4);
    moe_fc_gelu<<<g1, 128, 0, stream>>>(xb, wfcb, bfc + (size_t)e * IDIM, hbuf);
    cvt_to_bf16_v4<<<(nw4 + 255) / 256, 256, 0, stream>>>(
        wproj + (size_t)e * HDIM * IDIM, wprb, nw4);
    moe_proj_combine<<<g2, 128, 0, stream>>>(
        hbuf, wprb, bproj + (size_t)e * HDIM, coef, e, (e == 0) ? 1 : 0, out);
  }
  moe_finalize<<<1, 256, 0, stream>>>(probs, idxb, z2, aux);
}
